// MultiHeadAttention_83184926589395
// MI455X (gfx1250) — matrix operand order repaired and checked by reading, not compiled, not run on hardware
//
#include <hip/hip_runtime.h>
#include <stdint.h>

typedef uint16_t u16;
typedef uint32_t u32;
typedef __attribute__((ext_vector_type(16))) __bf16 v16bf;
typedef __attribute__((ext_vector_type(8)))  float  v8f;

#define B_  2
#define S_  2048
#define E_  1024
#define H_  16
#define DK  64

#define GAS __attribute__((address_space(1)))
#define LAS __attribute__((address_space(3)))

#if __has_builtin(__builtin_amdgcn_global_load_async_to_lds_b128)
#define ASYNC_LDS 1
#else
#define ASYNC_LDS 0
#endif

union Frag { v16bf v; uint4 q[2]; };

#if ASYNC_LDS
typedef __attribute__((__vector_size__(4 * sizeof(int)))) int i128v;
__device__ __forceinline__ void async_g2l_b128(const void* g, void* l) {
  __builtin_amdgcn_global_load_async_to_lds_b128(
      (GAS i128v*)(uintptr_t)g, (LAS i128v*)l, 0, 0);
}
#endif

__device__ __forceinline__ u16 f2bf(float x) {
  u32 u = __float_as_uint(x);
  u32 r = (u + 0x7FFFu + ((u >> 16) & 1u)) >> 16;
  return (u16)r;
}

__device__ __forceinline__ u32 pk2bf(float a, float b) {
#if __has_builtin(__builtin_amdgcn_cvt_pk_bf16_f32)
  typedef __attribute__((ext_vector_type(2))) __bf16 v2bf;
  union { v2bf v; u32 u; } cv;
  cv.v = __builtin_amdgcn_cvt_pk_bf16_f32(a, b);
  return cv.u;
#else
  return (u32)f2bf(a) | ((u32)f2bf(b) << 16);
#endif
}

__device__ __forceinline__ void wait_async0() {
#if __has_builtin(__builtin_amdgcn_s_wait_asynccnt)
  __builtin_amdgcn_s_wait_asynccnt(0);
#else
  asm volatile("s_wait_asynccnt 0x0" ::: "memory");
#endif
}

__device__ __forceinline__ v8f v8f_zero() {
  v8f z = {0.f,0.f,0.f,0.f,0.f,0.f,0.f,0.f};
  return z;
}

__device__ __forceinline__ v8f wmma_bf16(const Frag& a, const Frag& b, v8f c) {
  return __builtin_amdgcn_wmma_f32_16x16x32_bf16(false, a.v, false, b.v,
                                                 (short)0, c, false, false);
}

// A-matrix fragment (16x32 per kstep). ISA 16-bit A table:
// lane<16: K = {0..7, 16..23}; lane>=16: K = {8..15, 24..31} (u16 units).
__device__ __forceinline__ Frag frag_a(const u16* base, int strideH, int row, int ks, int lane) {
  int off = row * strideH + ks * 32 + (((lane & 31) >= 16) ? 8 : 0);
  Frag f;
  f.q[0] = *(const uint4*)(base + off);
  f.q[1] = *(const uint4*)(base + off + 16);
  return f;
}

// B-matrix fragment (32x16 per kstep) from N-major (K-contiguous) LDS tile.
// lane<16: K = {0..7, 16..23}; lane>=16: K = {8..15, 24..31}.
__device__ __forceinline__ Frag frag_b(const u16* base, int strideH, int col, int ks, int lane) {
  int off = col * strideH + ks * 32 + (((lane & 31) >= 16) ? 8 : 0);
  Frag f;
  f.q[0] = *(const uint4*)(base + off);
  f.q[1] = *(const uint4*)(base + off + 16);
  return f;
}

// ---------------------------------------------------------------------------
// Kernel 1: per-head projections  qp/kp/vp = x_h @ W_h + b_h  (bf16 out)
// ---------------------------------------------------------------------------
__device__ __forceinline__ void load_w(const float* __restrict__ Wg, u16* wTp, int t) {
  int d = (t * 16) >> 6, e0 = (t * 16) & 63;
  const float4* src = (const float4*)(Wg + t * 16);
  #pragma unroll
  for (int i = 0; i < 4; i++) {
    float4 f = src[i];
    wTp[(e0 + 4 * i + 0) * 64 + d] = f2bf(f.x);
    wTp[(e0 + 4 * i + 1) * 64 + d] = f2bf(f.y);
    wTp[(e0 + 4 * i + 2) * 64 + d] = f2bf(f.z);
    wTp[(e0 + 4 * i + 3) * 64 + d] = f2bf(f.w);
  }
}

__device__ __forceinline__ void load_x(const float* __restrict__ Xg, u16* xsp, int t) {
  int r = t >> 1, c0 = (t & 1) * 32;
  const float4* src = (const float4*)(Xg + (size_t)r * E_ + c0);
  uint4* dst = (uint4*)&xsp[r * 64 + c0];
  #pragma unroll
  for (int i = 0; i < 4; i++) {
    float4 f0 = src[2 * i], f1 = src[2 * i + 1];
    uint4 o;
    o.x = pk2bf(f0.x, f0.y); o.y = pk2bf(f0.z, f0.w);
    o.z = pk2bf(f1.x, f1.y); o.w = pk2bf(f1.z, f1.w);
    dst[i] = o;
  }
}

__device__ __forceinline__ void proj_tile(
    const u16* xsp, const u16* wTp, const float* __restrict__ biasp,
    u16* __restrict__ outp, u16* stage, int lane, int w, int h, size_t rowbase) {
  #pragma unroll
  for (int nt = 0; nt < 4; nt++) {
    v8f acc = v8f_zero();
    #pragma unroll
    for (int ks = 0; ks < 2; ks++) {
      Frag a  = frag_a(xsp, 64, w * 16 + (lane & 15), ks, lane);
      Frag bf = frag_b(wTp, 64, nt * 16 + (lane & 15), ks, lane);
      acc = wmma_bf16(a, bf, acc);
    }
    int e = nt * 16 + (lane & 15);
    float bias = biasp[h * DK + e];
    int rbase = w * 16 + ((lane >= 16) ? 8 : 0);
    #pragma unroll
    for (int j = 0; j < 8; j++)
      stage[(rbase + j) * 64 + e] = f2bf(acc[j] + bias);
  }
  asm volatile("" ::: "memory");  // same-wave LDS RAW; LDS pipe is in-order
  int r = w * 16 + (lane >> 1), c0 = (lane & 1) * 32;
  u16* dst = outp + ((rowbase + r) * DK + c0);
  const uint4* sq = (const uint4*)&stage[r * 64 + c0];
  uint4* dq = (uint4*)dst;
  dq[0] = sq[0]; dq[1] = sq[1]; dq[2] = sq[2]; dq[3] = sq[3];
  asm volatile("" ::: "memory");
}

__global__ void __launch_bounds__(256) proj_kernel(
    const float* __restrict__ q, const float* __restrict__ k, const float* __restrict__ v,
    const float* __restrict__ Wq, const float* __restrict__ bq,
    const float* __restrict__ Wk, const float* __restrict__ bk,
    const float* __restrict__ Wv, const float* __restrict__ bv,
    u16* __restrict__ Qp, u16* __restrict__ Kp, u16* __restrict__ Vp) {
  __shared__ __align__(16) u16 wTq[64 * 64], wTk[64 * 64], wTv[64 * 64];
  __shared__ __align__(16) u16 xsq[128 * 64], xsk[128 * 64], xsv[128 * 64];
  __shared__ __align__(16) u16 stage[128 * 64];

  const int stile = blockIdx.x, h = blockIdx.y, b = blockIdx.z;
  const int t = threadIdx.x, lane = t & 31, w = t >> 5;
  const int sbase = stile * 128;

  load_w(Wq + (size_t)h * DK * DK, wTq, t);
  load_w(Wk + (size_t)h * DK * DK, wTk, t);
  load_w(Wv + (size_t)h * DK * DK, wTv, t);
  const size_t xoff = ((size_t)b * S_ + sbase) * E_ + h * DK;
  load_x(q + xoff, xsq, t);
  load_x(k + xoff, xsk, t);
  load_x(v + xoff, xsv, t);
  __syncthreads();

  const size_t rowbase = (size_t)(b * H_ + h) * S_ + sbase;
  proj_tile(xsq, wTq, bq, Qp, stage, lane, w, h, rowbase);
  proj_tile(xsk, wTk, bk, Kp, stage, lane, w, h, rowbase);
  proj_tile(xsv, wTv, bv, Vp, stage, lane, w, h, rowbase);
}

// ---------------------------------------------------------------------------
// Kernel 2: flash attention per (b,h). Block = 128 query rows, 8 waves;
// each wave owns 16 query rows -> private online softmax, no cross-wave reduce.
// ---------------------------------------------------------------------------
__global__ void __launch_bounds__(256) attn_kernel(
    const u16* __restrict__ Qp, const u16* __restrict__ Kp, const u16* __restrict__ Vp,
    u16* __restrict__ O) {
  __shared__ __align__(16) u16 kt[32 * 64];      // K tile [key][dk]
  __shared__ __align__(16) u16 vt[64 * 32];      // V tile transposed [dk][key]
  __shared__ __align__(16) u16 pst[8][16 * 64];  // per-wave: P (16x32) / O-stage (16x64)

  const int qtile = blockIdx.x, h = blockIdx.y, b = blockIdx.z;
  const int t = threadIdx.x, lane = t & 31, w = t >> 5;
  const int qbase = qtile * 128;
  const size_t headoff = (size_t)(b * H_ + h) * S_ * DK;
  const u16* Qh = Qp + headoff;
  const u16* Kh = Kp + headoff;
  const u16* Vh = Vp + headoff;

  // Preload this wave's Q as two A-fragments (16x64) straight from global.
  Frag qf[2];
  {
    const u16* qrow = Qh + (size_t)(qbase + w * 16 + (lane & 15)) * DK;
    #pragma unroll
    for (int ks = 0; ks < 2; ks++) {
      int off = ks * 32 + ((lane >= 16) ? 8 : 0);
      qf[ks].q[0] = *(const uint4*)(qrow + off);
      qf[ks].q[1] = *(const uint4*)(qrow + off + 16);
    }
  }

  float mprev[8], lsum[8];
  v8f acc[4];
  #pragma unroll
  for (int j = 0; j < 8; j++) { mprev[j] = -__builtin_inff(); lsum[j] = 0.f; }
  #pragma unroll
  for (int nt = 0; nt < 4; nt++) acc[nt] = v8f_zero();

  const float scale = 1.0f / (float)DK;
  const int qrow0 = qbase + w * 16 + ((lane >= 16) ? 8 : 0);
  const int nkb = (qbase + 128) / 32;  // causal limit for this block

  for (int kb = 0; kb < nkb; kb++) {
    const int kbase = kb * 32;
    __syncthreads();
    {  // cooperative staging: K row-major (async if available), V transposed
      int key = t >> 3, dg = (t & 7) * 8;
      const u16* ksrc = Kh + (size_t)(kbase + key) * DK + dg;
#if ASYNC_LDS
      async_g2l_b128(ksrc, &kt[key * 64 + dg]);
#else
      *(uint4*)&kt[key * 64 + dg] = *(const uint4*)ksrc;
#endif
      uint4 vv = *(const uint4*)(Vh + (size_t)(kbase + key) * DK + dg);
      const u16* ve = (const u16*)&vv;
      #pragma unroll
      for (int i = 0; i < 8; i++) vt[(dg + i) * 32 + key] = ve[i];
      if (kb + 1 < nkb) {
        __builtin_prefetch(Kh + (size_t)(kbase + 32 + key) * DK + dg, 0, 1);
        __builtin_prefetch(Vh + (size_t)(kbase + 32 + key) * DK + dg, 0, 1);
      }
    }
#if ASYNC_LDS
    wait_async0();
#endif
    __syncthreads();

    // Scores: two 16x16 tiles (keys 0..15 / 16..31), K-dim 64 -> 2 WMMAs each
    v8f s0 = v8f_zero(), s1 = v8f_zero();
    #pragma unroll
    for (int ks = 0; ks < 2; ks++) {
      Frag b0 = frag_b(kt, 64, (lane & 15), ks, lane);
      Frag b1 = frag_b(kt, 64, 16 + (lane & 15), ks, lane);
      s0 = wmma_bf16(qf[ks], b0, s0);
      s1 = wmma_bf16(qf[ks], b1, s1);
    }

    const int kc0 = kbase + (lane & 15);
    const int kc1 = kc0 + 16;
    const float NEG = -__builtin_inff();
    #pragma unroll
    for (int j = 0; j < 8; j++) {
      int qr = qrow0 + j;
      float a0 = s0[j] * scale;
      float a1 = s1[j] * scale;
      if (a0 == 0.f) a0 = NEG;            // reference "padding mask"
      if (a1 == 0.f) a1 = NEG;
      if (kc0 > qr) a0 = NEG;             // causal mask
      if (kc1 > qr) a1 = NEG;
      float m = fmaxf(a0, a1);
      #pragma unroll
      for (int off = 1; off < 16; off <<= 1)   // row-max within 16-lane half
        m = fmaxf(m, __shfl_xor(m, off, 32));
      float mnew = fmaxf(mprev[j], m);
      float alpha, p0, p1;
      if (mnew == NEG) { alpha = 1.f; p0 = 0.f; p1 = 0.f; }
      else {
        alpha = __expf(mprev[j] - mnew);
        p0 = __expf(a0 - mnew);
        p1 = __expf(a1 - mnew);
      }
      float rs = p0 + p1;
      #pragma unroll
      for (int off = 1; off < 16; off <<= 1) rs += __shfl_xor(rs, off, 32);
      lsum[j] = lsum[j] * alpha + rs;
      mprev[j] = mnew;
      #pragma unroll
      for (int nt = 0; nt < 4; nt++) acc[nt][j] *= alpha;
      int prow = ((lane >= 16) ? 8 : 0) + j;
      pst[w][prow * 32 + (lane & 15)]      = f2bf(p0);   // C-layout -> A-layout
      pst[w][prow * 32 + 16 + (lane & 15)] = f2bf(p1);
    }
    asm volatile("" ::: "memory");

    Frag pf = frag_a(&pst[w][0], 32, lane & 15, 0, lane);
    #pragma unroll
    for (int nt = 0; nt < 4; nt++) {
      Frag bv = frag_b(vt, 32, nt * 16 + (lane & 15), 0, lane);
      acc[nt] = wmma_bf16(pf, bv, acc[nt]);
    }
    asm volatile("" ::: "memory");
  }

  // Finalize: 1/l (fully-masked rows -> 0), stage bf16, coalesced store.
  float inv[8];
  #pragma unroll
  for (int j = 0; j < 8; j++) inv[j] = (lsum[j] > 0.f) ? (1.f / lsum[j]) : 0.f;
  #pragma unroll
  for (int nt = 0; nt < 4; nt++) {
    int e = nt * 16 + (lane & 15);
    int rbase = (lane >= 16) ? 8 : 0;
    #pragma unroll
    for (int j = 0; j < 8; j++)
      pst[w][(rbase + j) * 64 + e] = f2bf(acc[nt][j] * inv[j]);
  }
  asm volatile("" ::: "memory");
  {
    int r = lane >> 1, c0 = (lane & 1) * 32;
    u16* dst = O + (((size_t)b * S_ + qbase + w * 16 + r) * E_ + h * DK + c0);
    const uint4* sq = (const uint4*)&pst[w][r * 64 + c0];
    uint4* dq = (uint4*)dst;
    dq[0] = sq[0]; dq[1] = sq[1]; dq[2] = sq[2]; dq[3] = sq[3];
  }
}

// ---------------------------------------------------------------------------
// Kernel 3: FC GEMM (4096x1024 @ 1024x1024) + bias + residual, f32 out
// ---------------------------------------------------------------------------
__global__ void __launch_bounds__(256) fc_kernel(
    const u16* __restrict__ O, const float* __restrict__ Wfc,
    const float* __restrict__ bfc, const float* __restrict__ vres,
    float* __restrict__ Y) {
  __shared__ __align__(16) u16 at[128 * 32];   // A tile [row][k]
  __shared__ __align__(16) u16 bt[64 * 32];    // Wfc tile transposed [n][k]
  __shared__ __align__(16) float yst[8][16 * 64];

  const int ntile = blockIdx.x, mtile = blockIdx.y;
  const int t = threadIdx.x, lane = t & 31, w = t >> 5;
  const int m0 = mtile * 128, n0 = ntile * 64;

  v8f acc[4];
  #pragma unroll
  for (int nt = 0; nt < 4; nt++) acc[nt] = v8f_zero();

  for (int kb = 0; kb < 32; kb++) {
    __syncthreads();
    {  // A: bf16 attention output (async global->LDS if available)
      int r = t >> 1, c0 = (t & 1) * 16;
      const u16* src = O + (size_t)(m0 + r) * E_ + kb * 32 + c0;
#if ASYNC_LDS
      async_g2l_b128(src,     &at[r * 32 + c0]);
      async_g2l_b128(src + 8, &at[r * 32 + c0 + 8]);
#else
      *(uint4*)&at[r * 32 + c0]     = ((const uint4*)src)[0];
      *(uint4*)&at[r * 32 + c0 + 8] = ((const uint4*)src)[1];
#endif
    }
    {  // B: f32 Wfc[k][n] -> bf16 transposed bt[n][k]
      int kk = t >> 3, ng = (t & 7) * 8;
      const float* src = Wfc + (size_t)(kb * 32 + kk) * E_ + n0 + ng;
      #pragma unroll
      for (int i = 0; i < 8; i++) bt[(ng + i) * 32 + kk] = f2bf(src[i]);
    }
#if ASYNC_LDS
    wait_async0();
#endif
    __syncthreads();
    Frag a = frag_a(at, 32, w * 16 + (lane & 15), 0, lane);
    #pragma unroll
    for (int nt = 0; nt < 4; nt++) {
      Frag bf = frag_b(bt, 32, nt * 16 + (lane & 15), 0, lane);
      acc[nt] = wmma_bf16(a, bf, acc[nt]);
    }
  }

  // Epilogue: stage f32, then coalesced bias + residual + store
  #pragma unroll
  for (int nt = 0; nt < 4; nt++) {
    int e = nt * 16 + (lane & 15);
    int rbase = (lane >= 16) ? 8 : 0;
    #pragma unroll
    for (int j = 0; j < 8; j++) yst[w][(rbase + j) * 64 + e] = acc[nt][j];
  }
  asm volatile("" ::: "memory");
  {
    int r = lane >> 1, c0 = (lane & 1) * 32;
    size_t m = (size_t)m0 + w * 16 + r;
    #pragma unroll
    for (int i = 0; i < 32; i += 4) {
      int n = n0 + c0 + i;
      float4 ys = *(const float4*)&yst[w][r * 64 + c0 + i];
      float4 vv = *(const float4*)(vres + m * E_ + n);
      float4 bb = *(const float4*)(bfc + n);
      float4 o;
      o.x = ys.x + vv.x + bb.x; o.y = ys.y + vv.y + bb.y;
      o.z = ys.z + vv.z + bb.z; o.w = ys.w + vv.w + bb.w;
      *(float4*)(Y + m * E_ + n) = o;
    }
  }
}

// ---------------------------------------------------------------------------
// Kernel 4: LayerNorm over the SEQUENCE axis (per (b,e) column), in place.
// ---------------------------------------------------------------------------
__global__ void __launch_bounds__(512) ln_kernel(
    float* __restrict__ Y, const float* __restrict__ gamma,
    const float* __restrict__ beta) {
  __shared__ float ssum[32][16];
  __shared__ float ssq[32][16];
  const int e = blockIdx.x * 16 + threadIdx.x;
  const int b = blockIdx.y;
  const int ty = threadIdx.y, tx = threadIdx.x;
  float* base = Y + (size_t)b * S_ * E_;

  float sum = 0.f, sq = 0.f;
  for (int s = ty; s < S_; s += 32) {
    float x = base[(size_t)s * E_ + e];
    sum += x; sq += x * x;
  }
  ssum[ty][tx] = sum; ssq[ty][tx] = sq;
  __syncthreads();
  for (int off = 16; off > 0; off >>= 1) {
    if (ty < off) { ssum[ty][tx] += ssum[ty + off][tx]; ssq[ty][tx] += ssq[ty + off][tx]; }
    __syncthreads();
  }
  float mean = ssum[0][tx] * (1.0f / S_);
  float var = ssq[0][tx] * (1.0f / S_) - mean * mean;
  float rstd = rsqrtf(var + 1e-4f);
  for (int s = ty; s < S_; s += 32) {
    size_t idx = (size_t)s * E_ + e;
    float x = base[idx];
    base[idx] = (x - mean) * rstd * gamma[s] + beta[s];
  }
}

// ---------------------------------------------------------------------------
extern "C" void kernel_launch(void* const* d_in, const int* in_sizes, int n_in,
                              void* d_out, int out_size, void* d_ws, size_t ws_size,
                              hipStream_t stream) {
  (void)in_sizes; (void)n_in; (void)out_size; (void)ws_size;
  const float* q     = (const float*)d_in[0];
  const float* k     = (const float*)d_in[1];
  const float* v     = (const float*)d_in[2];
  const float* Wq    = (const float*)d_in[3];
  const float* bq    = (const float*)d_in[4];
  const float* bk    = (const float*)d_in[6];
  const float* Wk    = (const float*)d_in[5];
  const float* Wv    = (const float*)d_in[7];
  const float* bv    = (const float*)d_in[8];
  const float* Wfc   = (const float*)d_in[9];
  const float* bfc   = (const float*)d_in[10];
  const float* gamma = (const float*)d_in[11];
  const float* beta  = (const float*)d_in[12];

  const size_t nElem = (size_t)B_ * H_ * S_ * DK;  // 4,194,304
  u16* Qp = (u16*)d_ws;
  u16* Kp = Qp + nElem;
  u16* Vp = Kp + nElem;
  u16* O  = Vp + nElem;
  float* Y = (float*)d_out;

  proj_kernel<<<dim3(S_ / 128, H_, B_), 256, 0, stream>>>(
      q, k, v, Wq, bq, Wk, bk, Wv, bv, Qp, Kp, Vp);
  attn_kernel<<<dim3(S_ / 128, H_, B_), 256, 0, stream>>>(Qp, Kp, Vp, O);
  fc_kernel<<<dim3(E_ / 64, (B_ * S_) / 128), 256, 0, stream>>>(O, Wfc, bfc, v, Y);
  ln_kernel<<<dim3(E_ / 16, B_), dim3(16, 32), 0, stream>>>(Y, gamma, beta);
}
